// Block_39462159516184
// MI455X (gfx1250) — compile-verified, matrix operand order checked by reading, not run on hardware
//
#include <hip/hip_runtime.h>
#include <hip/hip_bf16.h>
#include <math.h>

// Transformer block for MI455X (gfx1250, wave32, WMMA).
// All GEMMs (qkv, QK^T, P*V, proj, fc1, fc2) run on v_wmma_f32_16x16x32_bf16.
// Inner loops stage operands into double-buffered LDS tiles with
// GLOBAL_LOAD_ASYNC_TO_LDS_B128 (ASYNCcnt), so the DMA for K-step i+1 overlaps
// the WMMAs of step i; fragments are then read from LDS (ds_load_b128 pattern).

#define BATCH   4
#define SEQ     2048
#define MODEL   384
#define HEADS   6
#define HD      64
#define HIDDEN  1536
#define MTOT    (BATCH * SEQ)      // 8192 rows
#define ATT_SCALE 0.125f           // 64^-0.5

typedef __attribute__((ext_vector_type(16))) __bf16 v16bf;
typedef __attribute__((ext_vector_type(8)))  float  v8f;

union FragU { v16bf v; unsigned u[8]; };

__device__ __forceinline__ v8f wmma_bf16(v16bf a, v16bf b, v8f c) {
    // (neg_a, A, neg_b, B, c_mod, C, reuse_a, reuse_b)
    return __builtin_amdgcn_wmma_f32_16x16x32_bf16(false, a, false, b, (short)0, c, false, false);
}

// Per-lane async DMA: copy 16 bytes global -> LDS. VDST is the LDS byte address
// (low 32 bits of the generic LDS pointer, per ISA 10.2 aperture mapping).
__device__ __forceinline__ void async_copy16(const void* g, void* l) {
    asm volatile("global_load_async_to_lds_b128 %0, %1, off"
                 :: "v"((unsigned)(unsigned long)l), "v"(g)
                 : "memory");
}
__device__ __forceinline__ void wait_async_0()  { asm volatile("s_wait_asynccnt 0x0"  ::: "memory"); }
__device__ __forceinline__ void wait_async_12() { asm volatile("s_wait_asynccnt 0xc"  ::: "memory"); }
__device__ __forceinline__ void wait_async_16() { asm volatile("s_wait_asynccnt 0x10" ::: "memory"); }

// Stage an NROWS x 32-elem (or x EPR-elem) bf16 tile into LDS.
// Each issue moves 32 lanes x 16B = 512B. EPR = elements per row in LDS tile.
template<int EPR, int NROWS>
__device__ __forceinline__ void stage_tile(const __bf16* src, int ld, int row0, int k0,
                                           __bf16* dst, int lane) {
    constexpr int LPR = EPR / 8;        // lanes per row (16B = 8 bf16)
    constexpr int RPI = 32 / LPR;       // rows per issue
    constexpr int ISSUES = NROWS / RPI;
    int rr = lane / LPR;
    int cb = (lane % LPR) * 8;
#pragma unroll
    for (int t = 0; t < ISSUES; ++t) {
        int r = t * RPI + rr;
        async_copy16(src + (long)(row0 + r) * ld + k0 + cb, dst + r * EPR + cb);
    }
}

// 16x32 bf16 fragment from a row-major source with K contiguous (leading dim lda).
// CDNA5 16-bit A layout: lane holds row = lane%16; lanes 0-15: K 0-7 & 16-23,
// lanes 16-31: K 8-15 & 24-31. B operand of D=A*B mirrors it (lane holds column),
// so fragments of W^T / V^T / K loaded this way are exactly the B operand.
// Works on global or LDS pointers (addrspace inferred after inlining).
__device__ __forceinline__ v16bf load_frag(const __bf16* A, int lda, int row, int k0, int lane) {
    FragU f;
    int kb = (lane & 16) ? 8 : 0;
    const __bf16* p = A + (long)row * lda + k0 + kb;
#pragma unroll
    for (int m = 0; m < 4; ++m) {
        f.u[m]     = *(const unsigned*)(p + 2 * m);
        f.u[m + 4] = *(const unsigned*)(p + 16 + 2 * m);
    }
    return f.v;
}

// One wave computes a 32x64 f32 tile of A[M x K] @ W (W transposed, [Nc x K]).
// Double-buffered async LDS staging: 12 async issues per K-step (A 2KB + B 4KB),
// prefetch step i+1 while doing 8 WMMAs on step i.
__device__ __forceinline__ void gemm_tile_2x4(const __bf16* __restrict__ A,
                                              const __bf16* __restrict__ WT,
                                              int K, int tm, int tn, int lane,
                                              v8f acc[2][4],
                                              __bf16* sA, __bf16* sB) {
#pragma unroll
    for (int h = 0; h < 2; ++h)
#pragma unroll
        for (int nt = 0; nt < 4; ++nt) acc[h][nt] = (v8f){};
    const int nsteps = K / 32;
    stage_tile<32, 32>(A,  K, tm, 0, sA, lane);
    stage_tile<32, 64>(WT, K, tn, 0, sB, lane);
    for (int i = 0; i < nsteps; ++i) {
        int cur = i & 1;
        if (i + 1 < nsteps) {
            int nxt = cur ^ 1;
            stage_tile<32, 32>(A,  K, tm, (i + 1) * 32, sA + nxt * (32 * 32), lane);
            stage_tile<32, 64>(WT, K, tn, (i + 1) * 32, sB + nxt * (64 * 32), lane);
            wait_async_12();          // step i resident; step i+1 still in flight
        } else {
            wait_async_0();
        }
        const __bf16* at = sA + cur * (32 * 32);
        const __bf16* bt = sB + cur * (64 * 32);
        v16bf a0 = load_frag(at, 32, (lane & 15), 0, lane);
        v16bf a1 = load_frag(at, 32, 16 + (lane & 15), 0, lane);
#pragma unroll
        for (int nt = 0; nt < 4; ++nt) {
            v16bf b = load_frag(bt, 32, nt * 16 + (lane & 15), 0, lane);
            acc[0][nt] = wmma_bf16(a0, b, acc[0][nt]);
            acc[1][nt] = wmma_bf16(a1, b, acc[1][nt]);
        }
    }
}

// ---------------- utility kernels ----------------

// cast + transpose: src f32 [K][Nc] -> dst bf16 [Nc][K]
__global__ void cast_transpose_bf16(const float* __restrict__ src, __bf16* __restrict__ dst,
                                    int K, int Nc) {
    int i = blockIdx.x * blockDim.x + threadIdx.x;
    if (i < K * Nc) {
        int k = i / Nc, c = i % Nc;
        dst[(long)c * K + k] = (__bf16)src[i];
    }
}

// LayerNorm over 384 elems, one block (128 threads) per row, bf16 output.
__global__ __launch_bounds__(128) void ln_kernel(const float* __restrict__ x,
                                                 const float* __restrict__ g,
                                                 const float* __restrict__ be,
                                                 __bf16* __restrict__ y) {
    int row = blockIdx.x;
    int tid = threadIdx.x;
    const float* xr = x + (long)row * MODEL;
    float v0 = xr[tid], v1 = xr[tid + 128], v2 = xr[tid + 256];
    __shared__ float rs[128], rq[128];
    rs[tid] = v0 + v1 + v2;
    rq[tid] = v0 * v0 + v1 * v1 + v2 * v2;
    __syncthreads();
    for (int off = 64; off > 0; off >>= 1) {
        if (tid < off) { rs[tid] += rs[tid + off]; rq[tid] += rq[tid + off]; }
        __syncthreads();
    }
    float mean = rs[0] * (1.0f / MODEL);
    float var  = rq[0] * (1.0f / MODEL) - mean * mean;
    float inv  = rsqrtf(var + 1e-5f);
    __bf16* yr = y + (long)row * MODEL;
    yr[tid]       = (__bf16)((v0 - mean) * inv * g[tid]       + be[tid]);
    yr[tid + 128] = (__bf16)((v1 - mean) * inv * g[tid + 128] + be[tid + 128]);
    yr[tid + 256] = (__bf16)((v2 - mean) * inv * g[tid + 256] + be[tid + 256]);
}

// ---------------- GEMM kernels ----------------

// qkv = ln1 @ w_qkv; scatter into q/k [B,H,N,HD] and vT [B,H,HD,N] (bf16).
// tn is 64-aligned and heads are 64 wide, so which/head are uniform per block.
__global__ __launch_bounds__(32) void qkv_gemm(const __bf16* __restrict__ A,
                                               const __bf16* __restrict__ WT,
                                               __bf16* __restrict__ q,
                                               __bf16* __restrict__ k,
                                               __bf16* __restrict__ vT) {
    __shared__ __bf16 sA[2][32 * 32];
    __shared__ __bf16 sB[2][64 * 32];
    int lane = threadIdx.x;
    int tn = blockIdx.x * 64, tm = blockIdx.y * 32;
    v8f acc[2][4];
    gemm_tile_2x4(A, WT, MODEL, tm, tn, lane, acc, &sA[0][0], &sB[0][0]);
    int ro = (lane & 16) ? 8 : 0, col = lane & 15;
    int which = tn / MODEL;
    int head  = (tn % MODEL) >> 6;
#pragma unroll
    for (int hh = 0; hh < 2; ++hh)
#pragma unroll
        for (int nt = 0; nt < 4; ++nt)
#pragma unroll
            for (int r = 0; r < 8; ++r) {
                int gr = tm + hh * 16 + ro + r;
                int b = gr >> 11, n = gr & (SEQ - 1);
                int d = nt * 16 + col;
                __bf16 val = (__bf16)acc[hh][nt][r];
                if (which == 0)
                    q[(((long)b * HEADS + head) * SEQ + n) * HD + d] = val;
                else if (which == 1)
                    k[(((long)b * HEADS + head) * SEQ + n) * HD + d] = val;
                else
                    vT[(((long)b * HEADS + head) * HD + d) * SEQ + n] = val;
            }
}

// out(f32) = o @ w_proj + b_proj + x   (x1 residual)
__global__ __launch_bounds__(32) void proj_gemm(const __bf16* __restrict__ A,
                                                const __bf16* __restrict__ WT,
                                                const float* __restrict__ bias,
                                                const float* __restrict__ x,
                                                float* __restrict__ out) {
    __shared__ __bf16 sA[2][32 * 32];
    __shared__ __bf16 sB[2][64 * 32];
    int lane = threadIdx.x;
    int tn = blockIdx.x * 64, tm = blockIdx.y * 32;
    v8f acc[2][4];
    gemm_tile_2x4(A, WT, MODEL, tm, tn, lane, acc, &sA[0][0], &sB[0][0]);
    int ro = (lane & 16) ? 8 : 0, col = lane & 15;
#pragma unroll
    for (int hh = 0; hh < 2; ++hh)
#pragma unroll
        for (int nt = 0; nt < 4; ++nt)
#pragma unroll
            for (int r = 0; r < 8; ++r) {
                int gc = tn + nt * 16 + col;
                long idx = (long)(tm + hh * 16 + ro + r) * MODEL + gc;
                out[idx] = acc[hh][nt][r] + bias[gc] + x[idx];
            }
}

// h1(bf16) = gelu(ln2 @ w_fc1 + b_fc1)   (exact gelu, erf)
__global__ __launch_bounds__(32) void fc1_gemm(const __bf16* __restrict__ A,
                                               const __bf16* __restrict__ WT,
                                               const float* __restrict__ bias,
                                               __bf16* __restrict__ h1) {
    __shared__ __bf16 sA[2][32 * 32];
    __shared__ __bf16 sB[2][64 * 32];
    int lane = threadIdx.x;
    int tn = blockIdx.x * 64, tm = blockIdx.y * 32;
    v8f acc[2][4];
    gemm_tile_2x4(A, WT, MODEL, tm, tn, lane, acc, &sA[0][0], &sB[0][0]);
    int ro = (lane & 16) ? 8 : 0, col = lane & 15;
#pragma unroll
    for (int hh = 0; hh < 2; ++hh)
#pragma unroll
        for (int nt = 0; nt < 4; ++nt)
#pragma unroll
            for (int r = 0; r < 8; ++r) {
                int gc = tn + nt * 16 + col;
                float hv = acc[hh][nt][r] + bias[gc];
                float ge = 0.5f * hv * (1.0f + erff(hv * 0.70710678118f));
                h1[(long)(tm + hh * 16 + ro + r) * HIDDEN + gc] = (__bf16)ge;
            }
}

// out(f32) += h1 @ w_fc2 + b_fc2   (final residual; out already holds x1)
__global__ __launch_bounds__(32) void fc2_gemm(const __bf16* __restrict__ A,
                                               const __bf16* __restrict__ WT,
                                               const float* __restrict__ bias,
                                               float* __restrict__ out) {
    __shared__ __bf16 sA[2][32 * 32];
    __shared__ __bf16 sB[2][64 * 32];
    int lane = threadIdx.x;
    int tn = blockIdx.x * 64, tm = blockIdx.y * 32;
    v8f acc[2][4];
    gemm_tile_2x4(A, WT, HIDDEN, tm, tn, lane, acc, &sA[0][0], &sB[0][0]);
    int ro = (lane & 16) ? 8 : 0, col = lane & 15;
#pragma unroll
    for (int hh = 0; hh < 2; ++hh)
#pragma unroll
        for (int nt = 0; nt < 4; ++nt)
#pragma unroll
            for (int r = 0; r < 8; ++r) {
                int gc = tn + nt * 16 + col;
                long idx = (long)(tm + hh * 16 + ro + r) * MODEL + gc;
                out[idx] = out[idx] + acc[hh][nt][r] + bias[gc];
            }
}

// ---------------- flash attention ----------------
// One wave per (b, h, 16-query tile). KV streamed 32 columns at a time with
// double-buffered async LDS staging (K tile 32x64, V^T tile 64x32: 16 issues).
// S = Q K^T, online softmax with width-16 shuffles, P bounced through LDS
// (C-layout -> A-layout), then P*V from the V^T tile.
__global__ __launch_bounds__(32) void attn_kernel(const __bf16* __restrict__ qb,
                                                  const __bf16* __restrict__ kb,
                                                  const __bf16* __restrict__ vTb,
                                                  const float* __restrict__ mask,
                                                  __bf16* __restrict__ ob) {
    __shared__ __bf16 sK[2][32 * 64];
    __shared__ __bf16 sV[2][64 * 32];
    __shared__ __bf16 pl[16 * 32];
    int lane = threadIdx.x;
    int qt = blockIdx.x, h = blockIdx.y, b = blockIdx.z;
    const __bf16* Q  = qb  + ((long)b * HEADS + h) * SEQ * HD;
    const __bf16* Kh = kb  + ((long)b * HEADS + h) * SEQ * HD;
    const __bf16* VT = vTb + ((long)b * HEADS + h) * HD * SEQ;
    int ro = (lane & 16) ? 8 : 0, col = lane & 15;

    v16bf aq0 = load_frag(Q, HD, qt * 16 + (lane & 15), 0, lane);
    v16bf aq1 = load_frag(Q, HD, qt * 16 + (lane & 15), 32, lane);

    v8f o0 = {}, o1 = {}, o2 = {}, o3 = {};
    float m_i[8], l_i[8];
#pragma unroll
    for (int r = 0; r < 8; ++r) { m_i[r] = -1e30f; l_i[r] = 0.0f; }

    const float* Mbase = mask + ((long)b * SEQ + qt * 16) * SEQ;

    stage_tile<64, 32>(Kh, HD, 0, 0, &sK[0][0], lane);   // K rows 0..31, full 64 dims
    stage_tile<32, 64>(VT, SEQ, 0, 0, &sV[0][0], lane);  // V^T 64 dims x 32 seq

    for (int j = 0; j < SEQ; j += 32) {
        int cur = (j >> 5) & 1;
        if (j + 32 < SEQ) {
            int nxt = cur ^ 1;
            stage_tile<64, 32>(Kh, HD, j + 32, 0, &sK[nxt][0], lane);
            stage_tile<32, 64>(VT, SEQ, 0, j + 32, &sV[nxt][0], lane);
            wait_async_16();
        } else {
            wait_async_0();
        }
        const __bf16* kt = &sK[cur][0];
        const __bf16* vt = &sV[cur][0];

        v8f s0 = {}, s1 = {};
        s0 = wmma_bf16(aq0, load_frag(kt, 64, (lane & 15), 0, lane), s0);
        s0 = wmma_bf16(aq1, load_frag(kt, 64, (lane & 15), 32, lane), s0);
        s1 = wmma_bf16(aq0, load_frag(kt, 64, 16 + (lane & 15), 0, lane), s1);
        s1 = wmma_bf16(aq1, load_frag(kt, 64, 16 + (lane & 15), 32, lane), s1);

        float p0[8], p1[8];
#pragma unroll
        for (int r = 0; r < 8; ++r) {
            float sm0 = s0[r] * ATT_SCALE + Mbase[(long)(ro + r) * SEQ + j + col];
            float sm1 = s1[r] * ATT_SCALE + Mbase[(long)(ro + r) * SEQ + j + 16 + col];
            float t = fmaxf(sm0, sm1);
#pragma unroll
            for (int o = 1; o < 16; o <<= 1) t = fmaxf(t, __shfl_xor(t, o, 16));
            float mnew = fmaxf(m_i[r], t);
            float alpha = __expf(m_i[r] - mnew);
            float e0 = __expf(sm0 - mnew);
            float e1 = __expf(sm1 - mnew);
            float rs = e0 + e1;
#pragma unroll
            for (int o = 1; o < 16; o <<= 1) rs += __shfl_xor(rs, o, 16);
            l_i[r] = l_i[r] * alpha + rs;
            m_i[r] = mnew;
            o0[r] *= alpha; o1[r] *= alpha; o2[r] *= alpha; o3[r] *= alpha;
            p0[r] = e0; p1[r] = e1;
        }

        __syncthreads();
#pragma unroll
        for (int r = 0; r < 8; ++r) {
            pl[(ro + r) * 32 + col]      = (__bf16)p0[r];
            pl[(ro + r) * 32 + 16 + col] = (__bf16)p1[r];
        }
        __syncthreads();

        v16bf ap = load_frag(pl, 32, lane & 15, 0, lane);
        o0 = wmma_bf16(ap, load_frag(vt, 32,  0 + (lane & 15), 0, lane), o0);
        o1 = wmma_bf16(ap, load_frag(vt, 32, 16 + (lane & 15), 0, lane), o1);
        o2 = wmma_bf16(ap, load_frag(vt, 32, 32 + (lane & 15), 0, lane), o2);
        o3 = wmma_bf16(ap, load_frag(vt, 32, 48 + (lane & 15), 0, lane), o3);
    }

#pragma unroll
    for (int r = 0; r < 8; ++r) {
        float inv = 1.0f / l_i[r];
        long base = ((long)b * SEQ + qt * 16 + ro + r) * MODEL + h * HD;
        ob[base +  0 + col] = (__bf16)(o0[r] * inv);
        ob[base + 16 + col] = (__bf16)(o1[r] * inv);
        ob[base + 32 + col] = (__bf16)(o2[r] * inv);
        ob[base + 48 + col] = (__bf16)(o3[r] * inv);
    }
}

// ---------------- launch ----------------

extern "C" void kernel_launch(void* const* d_in, const int* in_sizes, int n_in,
                              void* d_out, int out_size, void* d_ws, size_t ws_size,
                              hipStream_t stream) {
    (void)in_sizes; (void)n_in; (void)out_size; (void)ws_size;
    const float* x      = (const float*)d_in[0];
    const float* mask   = (const float*)d_in[1];
    const float* w_qkv  = (const float*)d_in[2];
    const float* w_proj = (const float*)d_in[3];
    const float* b_proj = (const float*)d_in[4];
    const float* g1     = (const float*)d_in[5];
    const float* be1    = (const float*)d_in[6];
    const float* g2     = (const float*)d_in[7];
    const float* be2    = (const float*)d_in[8];
    const float* w_fc1  = (const float*)d_in[9];
    const float* b_fc1  = (const float*)d_in[10];
    const float* w_fc2  = (const float*)d_in[11];
    const float* b_fc2  = (const float*)d_in[12];
    float* out = (float*)d_out;

    char* ws = (char*)d_ws;
    size_t off = 0;
    auto alloc = [&](size_t bytes) {
        void* p = ws + off;
        off = (off + bytes + 255) & ~(size_t)255;
        return p;
    };
    __bf16* wqkvT  = (__bf16*)alloc(2ul * MODEL * 3 * MODEL);   // [1152][384]
    __bf16* wprojT = (__bf16*)alloc(2ul * MODEL * MODEL);       // [384][384]
    __bf16* wfc1T  = (__bf16*)alloc(2ul * MODEL * HIDDEN);      // [1536][384]
    __bf16* wfc2T  = (__bf16*)alloc(2ul * HIDDEN * MODEL);      // [384][1536]
    __bf16* ln1_b  = (__bf16*)alloc(2ul * MTOT * MODEL);
    __bf16* qbuf   = (__bf16*)alloc(2ul * MTOT * MODEL);
    __bf16* kbuf   = (__bf16*)alloc(2ul * MTOT * MODEL);
    __bf16* vTbuf  = (__bf16*)alloc(2ul * MTOT * MODEL);        // [B,H,HD,SEQ]
    __bf16* obuf   = (__bf16*)alloc(2ul * MTOT * MODEL);
    __bf16* ln2_b  = (__bf16*)alloc(2ul * MTOT * MODEL);
    __bf16* h1_b   = (__bf16*)alloc(2ul * MTOT * HIDDEN);

    // 1) weights -> bf16, transposed (L2-resident; trivial traffic)
    int n;
    n = MODEL * 3 * MODEL;
    cast_transpose_bf16<<<(n + 255) / 256, 256, 0, stream>>>(w_qkv, wqkvT, MODEL, 3 * MODEL);
    n = MODEL * MODEL;
    cast_transpose_bf16<<<(n + 255) / 256, 256, 0, stream>>>(w_proj, wprojT, MODEL, MODEL);
    n = MODEL * HIDDEN;
    cast_transpose_bf16<<<(n + 255) / 256, 256, 0, stream>>>(w_fc1, wfc1T, MODEL, HIDDEN);
    n = HIDDEN * MODEL;
    cast_transpose_bf16<<<(n + 255) / 256, 256, 0, stream>>>(w_fc2, wfc2T, HIDDEN, MODEL);

    // 2) LN1
    ln_kernel<<<MTOT, 128, 0, stream>>>(x, g1, be1, ln1_b);

    // 3) QKV gemm: [8192 x 384] @ [384 x 1152], 32x64 per wave
    qkv_gemm<<<dim3(3 * MODEL / 64, MTOT / 32), 32, 0, stream>>>(ln1_b, wqkvT, qbuf, kbuf, vTbuf);

    // 4) flash attention: (N/16, H, B) waves
    attn_kernel<<<dim3(SEQ / 16, HEADS, BATCH), 32, 0, stream>>>(qbuf, kbuf, vTbuf, mask, obuf);

    // 5) proj + residual -> out (x1, f32)
    proj_gemm<<<dim3(MODEL / 64, MTOT / 32), 32, 0, stream>>>(obuf, wprojT, b_proj, x, out);

    // 6) LN2 on x1
    ln_kernel<<<MTOT, 128, 0, stream>>>(out, g2, be2, ln2_b);

    // 7) fc1 + gelu
    fc1_gemm<<<dim3(HIDDEN / 64, MTOT / 32), 32, 0, stream>>>(ln2_b, wfc1T, b_fc1, h1_b);

    // 8) fc2 + residual into out
    fc2_gemm<<<dim3(MODEL / 64, MTOT / 32), 32, 0, stream>>>(h1_b, wfc2T, b_fc2, out);
}
